// RuleParse_17832704213028
// MI455X (gfx1250) — compile-verified, matrix operand order checked by reading, not run on hardware
//
#include <hip/hip_runtime.h>
#include <hip/hip_bf16.h>

// ---------------------------------------------------------------------------
// GATv2 x2 + mean-pool + FFN for MI455X (gfx1250, wave32, WMMA).
// Node features (<=40MB) are L2-resident (192MB L2); edge passes stream
// edge_attr; the 18xC edge matvec is fused into the score pass so no [E,C]
// intermediate exists. Dense GEMMs: LDS-staged bf16 WMMA, f32 accumulate.
// ---------------------------------------------------------------------------

#define NN   50000
#define EE   1600000
#define ETOT (EE + NN)     // edges + self loops
#define GG   512
#define NEGS 0.2f

typedef __attribute__((ext_vector_type(16))) __bf16        v16bf;
typedef __attribute__((ext_vector_type(8)))  float         v8f;
typedef __attribute__((ext_vector_type(8)))  unsigned int  v8u;
typedef __attribute__((ext_vector_type(4)))  unsigned int  u32x4;

#if __has_builtin(__builtin_amdgcn_global_load_async_to_lds_b32) && \
    __has_builtin(__builtin_amdgcn_s_wait_asynccnt)
#define HAVE_ASYNC_LDS 1
typedef __attribute__((address_space(1))) int GintT;   // global (AS1) int*
typedef __attribute__((address_space(3))) int LintT;   // LDS (AS3) int*
#endif

// order-preserving float<->uint encoding for atomicMax on signed floats
__device__ __forceinline__ unsigned int encf(float f) {
  unsigned int u = __float_as_uint(f);
  return u ^ ((u >> 31) ? 0xFFFFFFFFu : 0x80000000u);
}
__device__ __forceinline__ float decf(unsigned int u) {
  return __uint_as_float(u ^ ((u >> 31) ? 0x80000000u : 0xFFFFFFFFu));
}
#define ENC_NEG_INF 0x007FFFFFu   // encf(-inf)

__device__ __forceinline__ unsigned int pk_bf16(float lo, float hi) {
  __bf16 l = (__bf16)lo, h = (__bf16)hi;            // v_cvt_pk_bf16_f32
  unsigned short ul = __builtin_bit_cast(unsigned short, l);
  unsigned short uh = __builtin_bit_cast(unsigned short, h);
  return ((unsigned int)uh << 16) | ul;
}

// ---------------------------------------------------------------------------
// GEMM: C[M,Nc] = act(A[M,K] @ W[K,Nc] + bias). M % 16 == 0 for every call.
// Block = 256 thr (8 waves) -> 16(M) x 128(N) tile; K in chunks of 32.
// LDS holds packed-bf16 tiles; each wave builds its WMMA fragments with four
// aligned 16B LDS loads (ISA 16-bit A-layout K-pairs are contiguous per
// half-wave) and issues v_wmma_f32_16x16x32_bf16.
// ---------------------------------------------------------------------------
template <int K, int Nc>
__global__ void __launch_bounds__(256)
gemm_wmma_bf16(const float* __restrict__ A, int M,
               const float* __restrict__ W, const float* __restrict__ bias,
               float* __restrict__ Cout, int act)
{
  __shared__ unsigned int sA32[16 * 16];        // [m][kpair]  (16 pairs = 32 k)
  __shared__ unsigned int sBt32[128 * 20];      // [n][kpair], stride 20 (16B aligned)

  const int tid  = threadIdx.x;
  const int lane = tid & 31;
  const int wv   = tid >> 5;
  const int hi   = lane >> 4;
  const int nBlk = blockIdx.y * 128;
  const int nLoc = wv * 16 + (lane & 15);
  const int nG   = nBlk + nLoc;

  v8f c = {0.f, 0.f, 0.f, 0.f, 0.f, 0.f, 0.f, 0.f};
  constexpr int nCh = (K + 31) / 32;
#pragma unroll
  for (int ch = 0; ch < nCh; ++ch) {
    const int k0 = ch * 32;
    // ---- stage A tile: 256 pairs, one per thread (coalesced along k) ----
    {
      const int m   = tid >> 4;
      const int kk2 = tid & 15;
      const int k   = k0 + 2 * kk2;
      const long rb = (long)(blockIdx.x * 16 + m) * K;
      float f0 = (k     < K) ? A[rb + k]     : 0.f;
      float f1 = (k + 1 < K) ? A[rb + k + 1] : 0.f;
      sA32[tid] = pk_bf16(f0, f1);
    }
    // ---- stage W tile transposed: 2048 pairs, 8 per thread (coalesced in n) ----
#pragma unroll
    for (int it = 0; it < 8; ++it) {
      const int p   = tid + it * 256;
      const int n   = p & 127;
      const int kk2 = p >> 7;
      const int k   = k0 + 2 * kk2;
      const int ng  = nBlk + n;
      float f0 = (k     < K && ng < Nc) ? W[(long)k * Nc + ng]       : 0.f;
      float f1 = (k + 1 < K && ng < Nc) ? W[(long)(k + 1) * Nc + ng] : 0.f;
      sBt32[n * 20 + kk2] = pk_bf16(f0, f1);
    }
    __syncthreads();
    // ---- build fragments: two aligned 16B LDS loads per matrix ----
    const unsigned int* aRow = sA32 + (lane & 15) * 16;
    const unsigned int* bRow = sBt32 + nLoc * 20;
    u32x4 a0 = *(const u32x4*)(aRow + hi * 4);
    u32x4 a1 = *(const u32x4*)(aRow + 8 + hi * 4);
    u32x4 b0 = *(const u32x4*)(bRow + hi * 4);
    u32x4 b1 = *(const u32x4*)(bRow + 8 + hi * 4);
    v8u au = {a0.x, a0.y, a0.z, a0.w, a1.x, a1.y, a1.z, a1.w};
    v8u bu = {b0.x, b0.y, b0.z, b0.w, b1.x, b1.y, b1.z, b1.w};
    v16bf a = __builtin_bit_cast(v16bf, au);
    v16bf b = __builtin_bit_cast(v16bf, bu);
    c = __builtin_amdgcn_wmma_f32_16x16x32_bf16(false, a, false, b,
                                                (short)0, c, false, false);
    __syncthreads();
  }
  // ---- epilogue: C/D layout VGPR r -> M=r (+8 for hi half-wave) ----
  if (nG < Nc) {
    const float bn = bias[nG];
#pragma unroll
    for (int r = 0; r < 8; ++r) {
      int row = blockIdx.x * 16 + r + hi * 8;
      float v = c[r] + bn;
      if (act) v = fmaxf(v, 0.f);
      Cout[(long)row * Nc + nG] = v;
    }
  }
}

// ---------------------------------------------------------------------------
// Self-loop edge_attr = mean of incoming edge_attr (fill_value='mean')
// ---------------------------------------------------------------------------
__global__ void loop_accum(const int* __restrict__ dst, const float* __restrict__ ea,
                           float* __restrict__ loopSum, float* __restrict__ deg)
{
  long idx = (long)blockIdx.x * blockDim.x + threadIdx.x;
  if (idx >= (long)EE * 18) return;
  long e = idx / 18; int j = (int)(idx % 18);
  int d = dst[e];
  if (j == 0) atomicAdd(&deg[d], 1.0f);
  atomicAdd(&loopSum[(long)d * 18 + j], ea[idx]);
}
__global__ void loop_finish(float* __restrict__ loopSum, const float* __restrict__ deg)
{
  long i = (long)blockIdx.x * blockDim.x + threadIdx.x;
  if (i >= (long)NN * 18) return;
  loopSum[i] = loopSum[i] / fmaxf(deg[i / 18], 1.0f);
}

// ---------------------------------------------------------------------------
// Pass A: s_e = att . leaky_relu(xl[src] + xr[dst] + ea_e @ We); atomicMax m[dst]
// One wave per edge; We (18xC) + att staged in LDS via gfx1250
// global_load_async_to_lds (ASYNCcnt) when the toolchain exposes it.
// ---------------------------------------------------------------------------
__global__ void __launch_bounds__(256)
gat_edge_score(const float* __restrict__ xl, const float* __restrict__ xr,
               const float* __restrict__ We, const float* __restrict__ att,
               const float* __restrict__ ea, const float* __restrict__ eaLoop,
               const int* __restrict__ src, const int* __restrict__ dst,
               float* __restrict__ sOut, unsigned int* __restrict__ mEnc, int C)
{
  __shared__ float sW[18 * 200];
  __shared__ float sAtt[200];
#if HAVE_ASYNC_LDS
  for (int i = threadIdx.x; i < 18 * C; i += blockDim.x)
    __builtin_amdgcn_global_load_async_to_lds_b32(
        (GintT*)(uintptr_t)(We + i), (LintT*)(uintptr_t)(&sW[i]), 0, 0);
  for (int i = threadIdx.x; i < C; i += blockDim.x)
    __builtin_amdgcn_global_load_async_to_lds_b32(
        (GintT*)(uintptr_t)(att + i), (LintT*)(uintptr_t)(&sAtt[i]), 0, 0);
  __builtin_amdgcn_s_wait_asynccnt(0);
#else
  for (int i = threadIdx.x; i < 18 * C; i += blockDim.x) sW[i] = We[i];
  for (int i = threadIdx.x; i < C;      i += blockDim.x) sAtt[i] = att[i];
#endif
  __syncthreads();

  const int lane = threadIdx.x & 31;
  long e = (long)blockIdx.x * 8 + (threadIdx.x >> 5);
  if (e >= ETOT) return;

  int sN, dN; const float* eap;
  if (e < EE) { sN = src[e]; dN = dst[e]; eap = ea + e * 18; }
  else        { sN = dN = (int)(e - EE);  eap = eaLoop + (long)sN * 18; }

  float er[18];
#pragma unroll
  for (int j = 0; j < 18; ++j) er[j] = eap[j];   // broadcast load, L2 hit

  const float* xls = xl + (long)sN * C;
  const float* xrd = xr + (long)dN * C;
  float acc = 0.f;
  for (int f = lane; f < C; f += 32) {
    float g = 0.f;
#pragma unroll
    for (int j = 0; j < 18; ++j) g = fmaf(er[j], sW[j * C + f], g);
    float v = xls[f] + xrd[f] + g;
    v = (v > 0.f) ? v : NEGS * v;
    acc = fmaf(v, sAtt[f], acc);
  }
#pragma unroll
  for (int off = 16; off > 0; off >>= 1) acc += __shfl_xor(acc, off, 32);
  if (lane == 0) {
    sOut[e] = acc;
    atomicMax(&mEnc[dN], encf(acc));
  }
}

// Pass B: e_e = exp(s_e - m[dst]); z[dst] += e_e
__global__ void gat_edge_exp(const int* __restrict__ dst, float* __restrict__ s,
                             const unsigned int* __restrict__ mEnc, float* __restrict__ z)
{
  long e = (long)blockIdx.x * blockDim.x + threadIdx.x;
  if (e >= ETOT) return;
  int d = (e < EE) ? dst[e] : (int)(e - EE);
  float ev = __expf(s[e] - decf(mEnc[d]));
  s[e] = ev;
  atomicAdd(&z[d], ev);
}

// Pass C: out[dst] += (e_e / z[dst]) * xl[src]   (L2-resident scatter)
__global__ void __launch_bounds__(256)
gat_edge_aggregate(const float* __restrict__ xl, const int* __restrict__ src,
                   const int* __restrict__ dst, const float* __restrict__ s,
                   const float* __restrict__ z, float* __restrict__ out, int C)
{
  const int lane = threadIdx.x & 31;
  long e = (long)blockIdx.x * 8 + (threadIdx.x >> 5);
  if (e >= ETOT) return;
  int sN, dN;
  if (e < EE) { sN = src[e]; dN = dst[e]; } else { sN = dN = (int)(e - EE); }
  float alpha = s[e] / z[dN];
  const float* xrow = xl + (long)sN * C;
  float* orow = out + (long)dN * C;
  __builtin_prefetch(xrow, 0, 0);
  for (int f = lane; f < C; f += 32) atomicAdd(&orow[f], alpha * xrow[f]);
}

__global__ void bias_act(float* __restrict__ x, const float* __restrict__ bias,
                         long M, int C, int act)
{
  long i = (long)blockIdx.x * blockDim.x + threadIdx.x;
  if (i >= M * C) return;
  float v = x[i] + bias[i % C];
  x[i] = act ? fmaxf(v, 0.f) : v;
}

__global__ void __launch_bounds__(256)
pool_accum(const float* __restrict__ h3, const int* __restrict__ batch,
           float* __restrict__ pooled, float* __restrict__ cnt)
{
  const int lane = threadIdx.x & 31;
  long i = (long)blockIdx.x * 8 + (threadIdx.x >> 5);
  if (i >= NN) return;
  int b = batch[i];
  if (lane == 0) atomicAdd(&cnt[b], 1.0f);
  const float* row = h3 + i * 400;
  float* prow = pooled + (long)b * 400;
  for (int f = lane; f < 400; f += 32) atomicAdd(&prow[f], row[f]);
}
__global__ void pool_div(float* __restrict__ pooled, const float* __restrict__ cnt)
{
  long i = (long)blockIdx.x * blockDim.x + threadIdx.x;
  if (i >= (long)GG * 400) return;
  pooled[i] /= fmaxf(cnt[i / 400], 1.0f);
}

__global__ void fill_f32(float* p, float v, long n)
{ long i = (long)blockIdx.x * blockDim.x + threadIdx.x; if (i < n) p[i] = v; }
__global__ void fill_u32(unsigned int* p, unsigned int v, long n)
{ long i = (long)blockIdx.x * blockDim.x + threadIdx.x; if (i < n) p[i] = v; }

// ---------------------------------------------------------------------------
static inline unsigned int cdiv(long a, long b) { return (unsigned int)((a + b - 1) / b); }

extern "C" void kernel_launch(void* const* d_in, const int* in_sizes, int n_in,
                              void* d_out, int out_size, void* d_ws, size_t ws_size,
                              hipStream_t stream)
{
  const float* x    = (const float*)d_in[0];
  const int*   ei   = (const int*)  d_in[1];
  const int*   src  = ei;
  const int*   dst  = ei + EE;
  const float* ea   = (const float*)d_in[2];
  const int*   bat  = (const int*)  d_in[3];
  const float* W1l  = (const float*)d_in[4];  const float* b1l = (const float*)d_in[5];
  const float* W1r  = (const float*)d_in[6];  const float* b1r = (const float*)d_in[7];
  const float* W1e  = (const float*)d_in[8];  const float* a1  = (const float*)d_in[9];
  const float* c1   = (const float*)d_in[10];
  const float* W2l  = (const float*)d_in[11]; const float* b2l = (const float*)d_in[12];
  const float* W2r  = (const float*)d_in[13]; const float* b2r = (const float*)d_in[14];
  const float* W2e  = (const float*)d_in[15]; const float* a2  = (const float*)d_in[16];
  const float* c2   = (const float*)d_in[17];
  const float* W3   = (const float*)d_in[18]; const float* b3  = (const float*)d_in[19];
  const float* F1   = (const float*)d_in[20]; const float* bf1 = (const float*)d_in[21];
  const float* F2   = (const float*)d_in[22]; const float* bf2 = (const float*)d_in[23];
  const float* F3   = (const float*)d_in[24]; const float* bf3 = (const float*)d_in[25];
  float* out = (float*)d_out;
  float* ws  = (float*)d_ws;

  // workspace layout (floats), with reuse: xl2 overlays xl1+xr1; pooling
  // buffers overlay the dead score/m/z region. Peak ~57.7M floats (~231MB).
  float*        loopEa = ws;                            //  900000
  float*        deg    = ws + 900000;                   //   50000
  float*        sbuf   = ws + 950000;                   // 1650000
  unsigned int* mEnc   = (unsigned int*)(ws + 2600000); //   50000
  float*        z      = ws + 2650000;                  //   50000
  float*        xl1    = ws + 2700000;                  // 5000000
  float*        xr1    = ws + 7700000;                  // 5000000
  float*        h1     = ws + 12700000;                 // 5000000
  float*        xl2    = ws + 2700000;                  // reuse xl1+xr1 (10M)
  float*        xr2    = ws + 17700000;                 // 10000000
  float*        out2   = ws + 27700000;                 // 10000000
  float*        h3     = ws + 37700000;                 // 20000000 -> 57.7M
  float*        pooled = ws + 950000;                   // reuse (GG*400)
  float*        cnt    = pooled + (long)GG * 400;       // GG
  float*        y1     = cnt + GG;                      // GG*200
  float*        y2     = y1 + (long)GG * 200;           // GG*100

  const dim3 b256(256);

  // ---- self-loop edge_attr (mean of incoming) ----
  fill_f32<<<cdiv((long)NN * 18, 256), b256, 0, stream>>>(loopEa, 0.f, (long)NN * 18);
  fill_f32<<<cdiv(NN, 256), b256, 0, stream>>>(deg, 0.f, NN);
  loop_accum<<<cdiv((long)EE * 18, 256), b256, 0, stream>>>(dst, ea, loopEa, deg);
  loop_finish<<<cdiv((long)NN * 18, 256), b256, 0, stream>>>(loopEa, deg);

  // ---- layer 1: GATv2(16 -> 100) ----
  gemm_wmma_bf16<16, 100><<<dim3(NN / 16, 1), b256, 0, stream>>>(x, NN, W1l, b1l, xl1, 0);
  gemm_wmma_bf16<16, 100><<<dim3(NN / 16, 1), b256, 0, stream>>>(x, NN, W1r, b1r, xr1, 0);
  fill_u32<<<cdiv(NN, 256), b256, 0, stream>>>(mEnc, ENC_NEG_INF, NN);
  fill_f32<<<cdiv(NN, 256), b256, 0, stream>>>(z, 0.f, NN);
  gat_edge_score<<<cdiv(ETOT, 8), b256, 0, stream>>>(xl1, xr1, W1e, a1, ea, loopEa, src, dst, sbuf, mEnc, 100);
  gat_edge_exp<<<cdiv(ETOT, 256), b256, 0, stream>>>(dst, sbuf, mEnc, z);
  fill_f32<<<cdiv((long)NN * 100, 256), b256, 0, stream>>>(h1, 0.f, (long)NN * 100);
  gat_edge_aggregate<<<cdiv(ETOT, 8), b256, 0, stream>>>(xl1, src, dst, sbuf, z, h1, 100);
  bias_act<<<cdiv((long)NN * 100, 256), b256, 0, stream>>>(h1, c1, NN, 100, 1);

  // ---- layer 2: GATv2(100 -> 200) ----
  gemm_wmma_bf16<100, 200><<<dim3(NN / 16, 2), b256, 0, stream>>>(h1, NN, W2l, b2l, xl2, 0);
  gemm_wmma_bf16<100, 200><<<dim3(NN / 16, 2), b256, 0, stream>>>(h1, NN, W2r, b2r, xr2, 0);
  fill_u32<<<cdiv(NN, 256), b256, 0, stream>>>(mEnc, ENC_NEG_INF, NN);
  fill_f32<<<cdiv(NN, 256), b256, 0, stream>>>(z, 0.f, NN);
  gat_edge_score<<<cdiv(ETOT, 8), b256, 0, stream>>>(xl2, xr2, W2e, a2, ea, loopEa, src, dst, sbuf, mEnc, 200);
  gat_edge_exp<<<cdiv(ETOT, 256), b256, 0, stream>>>(dst, sbuf, mEnc, z);
  fill_f32<<<cdiv((long)NN * 200, 256), b256, 0, stream>>>(out2, 0.f, (long)NN * 200);
  gat_edge_aggregate<<<cdiv(ETOT, 8), b256, 0, stream>>>(xl2, src, dst, sbuf, z, out2, 200);
  bias_act<<<cdiv((long)NN * 200, 256), b256, 0, stream>>>(out2, c2, NN, 200, 1);

  // ---- linear 200 -> 400 ----
  gemm_wmma_bf16<200, 400><<<dim3(NN / 16, 4), b256, 0, stream>>>(out2, NN, W3, b3, h3, 0);

  // ---- mean pool over graphs ----
  fill_f32<<<cdiv((long)GG * 400, 256), b256, 0, stream>>>(pooled, 0.f, (long)GG * 400);
  fill_f32<<<cdiv(GG, 256), b256, 0, stream>>>(cnt, 0.f, GG);
  pool_accum<<<cdiv(NN, 8), b256, 0, stream>>>(h3, bat, pooled, cnt);
  pool_div<<<cdiv((long)GG * 400, 256), b256, 0, stream>>>(pooled, cnt);

  // ---- FFN 400 -> 200 -> 100 -> 100 ----
  gemm_wmma_bf16<400, 200><<<dim3(GG / 16, 2), b256, 0, stream>>>(pooled, GG, F1, bf1, y1, 1);
  gemm_wmma_bf16<200, 100><<<dim3(GG / 16, 1), b256, 0, stream>>>(y1, GG, F2, bf2, y2, 1);
  gemm_wmma_bf16<100, 100><<<dim3(GG / 16, 1), b256, 0, stream>>>(y2, GG, F3, bf3, out, 0);
}
